// HybridMemory_20169166422300
// MI455X (gfx1250) — compile-verified, matrix operand order checked by reading, not run on hardware
//
#include <hip/hip_runtime.h>
#include <hip/hip_bf16.h>

// Problem constants (from reference)
#define B_DIM 256
#define D_DIM 256
#define N_DIM 131072
#define INV_TEMP 20.0f     // 1 / 0.05
#define EPS 1e-6f

#define NTILES (N_DIM / 16)    // 8192 n-tiles of 16 columns
#define GRID_MAIN 512
#define NITER (NTILES / GRID_MAIN)   // 16 tiles per workgroup, contiguous

typedef __attribute__((ext_vector_type(16))) __bf16 v16bf;
typedef __attribute__((ext_vector_type(8)))  float  v8f;

// round-to-nearest-even float -> bf16 bits
static __device__ __forceinline__ unsigned short f2bf(float f) {
    union { float f; unsigned u; } v; v.f = f;
    unsigned u = v.u;
    u += 0x7fffu + ((u >> 16) & 1u);
    return (unsigned short)(u >> 16);
}

// ---------------------------------------------------------------------------
// Kernel 1: normalize inputs, emit x as bf16 in WMMA A-operand layout,
// compute picked logits x[b]·features[labels[b]] * INV_TEMP (fp32 exact),
// and zero the accumulators.
// A-layout (16-bit A 16x32, ISA 7.12.2): for m-tile mt, k-chunk kc:
//   lane L (0..31): hi = L>>4, row = L&15; 16 bf16 = K{kc*32+hi*8+0..7,
//   kc*32+16+hi*8+0..7}. Stored flat: elem = mt*4096 + kc*512 + L*16 + j.
// ---------------------------------------------------------------------------
__global__ __launch_bounds__(256) void prep_kernel(
    const float* __restrict__ inputs, const int* __restrict__ labels,
    const float* __restrict__ feats, unsigned short* __restrict__ xA,
    float* __restrict__ picked, float* __restrict__ S, float* __restrict__ Tsum)
{
    __shared__ float red[256];
    const int b = blockIdx.x, t = threadIdx.x;

    float v = inputs[b * D_DIM + t];
    red[t] = v * v;
    __syncthreads();
    for (int s = 128; s > 0; s >>= 1) {
        if (t < s) red[t] += red[t + s];
        __syncthreads();
    }
    float norm = fmaxf(sqrtf(red[0]), 1e-12f);
    float x = v / norm;

    // scatter into A layout
    const int mt = b >> 4, nl = b & 15;
    const int kc = t >> 5, r = t & 31;
    const int hi = (r >> 3) & 1;
    const int j  = (r & 7) + ((r & 16) ? 8 : 0);
    xA[mt * 4096 + kc * 512 + (hi * 16 + nl) * 16 + j] = f2bf(x);

    // picked logit (fp32 exact)
    const int lb = labels[b];
    float p = x * feats[(size_t)lb * D_DIM + t];
    __syncthreads();
    red[t] = p;
    __syncthreads();
    for (int s = 128; s > 0; s >>= 1) {
        if (t < s) red[t] += red[t + s];
        __syncthreads();
    }
    if (t == 0) {
        picked[b] = red[0] * INV_TEMP;
        S[b] = 0.0f;
        if (b == 0) *Tsum = 0.0f;
    }
}

// ---------------------------------------------------------------------------
// Kernel 2: the streaming GEMM-reduce. 512 threads = 16 waves; wave w owns
// M-tile w (rows 16w..16w+15). Features tiles (16 x 256 fp32) are loaded,
// converted to bf16 and stored to LDS pre-swizzled into the B-operand lane
// layout so compute reads are contiguous 32B/lane. Double buffered.
// Accumulates S[b] = sum_n exp(logit) and T = sum logits.
// ---------------------------------------------------------------------------
__global__ __launch_bounds__(512) void main_kernel(
    const float* __restrict__ feats, const unsigned short* __restrict__ xA,
    float* __restrict__ S, float* __restrict__ Tsum)
{
    // two buffers, each 8 k-chunks * 32 lanes * 16 bf16 = 4096 ushort = 8KB
    __shared__ alignas(32) unsigned short lbuf[2][4096];

    const int tid  = threadIdx.x;
    const int wave = tid >> 5;
    const int lane = tid & 31;

    // ---- load this wave's A operand (resident for whole kernel) ----
    v16bf a[8];
    {
        const unsigned short* ab = xA + wave * 4096 + lane * 16;
#pragma unroll
        for (int kc = 0; kc < 8; ++kc)
            a[kc] = *(const v16bf*)(ab + kc * 512);
    }

    // ---- loader thread mapping ----
    const int n_local = tid >> 5;          // 0..15 (row within tile)
    const int d0      = (tid & 31) * 8;    // 8 consecutive d per thread
    const int kcL = d0 >> 5, rL = d0 & 31;
    const int hiL = (rL >> 3) & 1;
    const int jbL = (rL & 16) ? 8 : 0;
    const int dstIdx = kcL * 512 + (hiL * 16 + n_local) * 16 + jbL; // ushort idx

    const int nt0 = blockIdx.x * NITER;

    float eacc[8];
#pragma unroll
    for (int i = 0; i < 8; ++i) eacc[i] = 0.0f;
    float tacc = 0.0f;

    // prologue: stage tile nt0 into buffer 0
    {
        const float* src = feats + ((size_t)(nt0 * 16 + n_local)) * D_DIM + d0;
        float4 f0 = *(const float4*)(src);
        float4 f1 = *(const float4*)(src + 4);
        uint4 p;
        p.x = (unsigned)f2bf(f0.x) | ((unsigned)f2bf(f0.y) << 16);
        p.y = (unsigned)f2bf(f0.z) | ((unsigned)f2bf(f0.w) << 16);
        p.z = (unsigned)f2bf(f1.x) | ((unsigned)f2bf(f1.y) << 16);
        p.w = (unsigned)f2bf(f1.z) | ((unsigned)f2bf(f1.w) << 16);
        *(uint4*)&lbuf[0][dstIdx] = p;
    }
    __syncthreads();

    for (int it = 0; it < NITER; ++it) {
        const int cur  = it & 1;
        const bool have = (it + 1) < NITER;

        // issue next tile's global loads before compute (latency overlap)
        uint4 pn;
        if (have) {
            const float* src =
                feats + ((size_t)((nt0 + it + 1) * 16 + n_local)) * D_DIM + d0;
            float4 f0 = *(const float4*)(src);
            float4 f1 = *(const float4*)(src + 4);
            pn.x = (unsigned)f2bf(f0.x) | ((unsigned)f2bf(f0.y) << 16);
            pn.y = (unsigned)f2bf(f0.z) | ((unsigned)f2bf(f0.w) << 16);
            pn.z = (unsigned)f2bf(f1.x) | ((unsigned)f2bf(f1.y) << 16);
            pn.w = (unsigned)f2bf(f1.z) | ((unsigned)f2bf(f1.w) << 16);
            if ((it + 2) < NITER) {
                // emits global_prefetch_b8 for the tile after next
                __builtin_prefetch(
                    feats + ((size_t)((nt0 + it + 2) * 16 + n_local)) * D_DIM + d0,
                    0, 1);
            }
        }

        // ---- compute: 8 WMMAs over K=256 for this (m-tile, n-tile) ----
        const unsigned short* bbase = &lbuf[cur][lane * 16];
        v8f c = {};
#pragma unroll
        for (int kc = 0; kc < 8; ++kc) {
            v16bf bb = *(const v16bf*)(bbase + kc * 512);
            c = __builtin_amdgcn_wmma_f32_16x16x32_bf16(
                    false, a[kc], false, bb, (short)0, c, false, false);
        }
#pragma unroll
        for (int i = 0; i < 8; ++i) {
            float l = c[i] * INV_TEMP;
            tacc += l;
            eacc[i] += __expf(l);
        }
        __syncthreads();

        if (have) {
            *(uint4*)&lbuf[1 - cur][dstIdx] = pn;
            __syncthreads();
        }
    }

    // ---- epilogue: reduce exp-partials across the 16-lane halves ----
    // lane group 0-15 holds rows wave*16 + i, group 16-31 rows wave*16 + 8 + i
#pragma unroll
    for (int i = 0; i < 8; ++i) {
        float v = eacc[i];
        v += __shfl_xor(v, 1, 32);
        v += __shfl_xor(v, 2, 32);
        v += __shfl_xor(v, 4, 32);
        v += __shfl_xor(v, 8, 32);
        if ((lane & 15) == 0) {
            int row = wave * 16 + ((lane >> 4) ? (8 + i) : i);
            atomicAdd(&S[row], v);
        }
    }
    // total logit sum across the full wave
    tacc += __shfl_xor(tacc, 1, 32);
    tacc += __shfl_xor(tacc, 2, 32);
    tacc += __shfl_xor(tacc, 4, 32);
    tacc += __shfl_xor(tacc, 8, 32);
    tacc += __shfl_xor(tacc, 16, 32);
    if (lane == 0) atomicAdd(Tsum, tacc);
}

// ---------------------------------------------------------------------------
// Kernel 3: softmax/log/NLL epilogue -> scalar loss.
// ---------------------------------------------------------------------------
__global__ __launch_bounds__(256) void finish_kernel(
    const float* __restrict__ S, const float* __restrict__ picked,
    const float* __restrict__ Tsum, float* __restrict__ out)
{
    __shared__ float red[256];
    const int t = threadIdx.x;
    const float m = *Tsum * (1.0f / ((float)B_DIM * (float)N_DIM));
    // masked_sums = exp(-m) * S[b] + EPS ; masked_sim = exp(l_b - m)/masked_sums
    float denom = __expf(-m) * S[t] + EPS;
    float prob  = __expf(picked[t] - m) / denom;
    float lp    = __logf(prob + EPS);
    red[t] = lp;
    __syncthreads();
    for (int s = 128; s > 0; s >>= 1) {
        if (t < s) red[t] += red[t + s];
        __syncthreads();
    }
    if (t == 0) out[0] = -red[0] / (float)B_DIM;
}

// ---------------------------------------------------------------------------
extern "C" void kernel_launch(void* const* d_in, const int* in_sizes, int n_in,
                              void* d_out, int out_size, void* d_ws, size_t ws_size,
                              hipStream_t stream) {
    const float* inputs   = (const float*)d_in[0];   // [256,256] f32
    const int*   labels   = (const int*)d_in[1];     // [256] i32
    // d_in[2] (index) and d_in[3] (average_center) do not enter forward math
    const float* features = (const float*)d_in[4];   // [131072,256] f32

    char* ws = (char*)d_ws;
    unsigned short* xA    = (unsigned short*)(ws);            // 131072 B
    float*          picked = (float*)(ws + 131072);           // 1024 B
    float*          S      = (float*)(ws + 132096);           // 1024 B
    float*          Tsum   = (float*)(ws + 133120);           // 4 B

    prep_kernel<<<B_DIM, 256, 0, stream>>>(inputs, labels, features,
                                           xA, picked, S, Tsum);
    main_kernel<<<GRID_MAIN, 512, 0, stream>>>(features, xA, S, Tsum);
    finish_kernel<<<1, 256, 0, stream>>>(S, picked, Tsum, (float*)d_out);
}
